// FeatureMatching_52982716563770
// MI455X (gfx1250) — matrix operand order repaired and checked by reading, not compiled, not run on hardware
//
#include <hip/hip_runtime.h>
#include <hip/hip_bf16.h>

typedef __attribute__((ext_vector_type(16))) _Float16 v16h;
typedef __attribute__((ext_vector_type(8)))  float    v8f;
typedef __attribute__((ext_vector_type(2)))  _Float16 h2;
typedef __attribute__((ext_vector_type(4)))  unsigned int v4u;
typedef __attribute__((ext_vector_type(8)))  int v8i;
typedef __attribute__((ext_vector_type(4)))  int v4i;

#if __has_builtin(__builtin_amdgcn_tensor_load_to_lds) && __has_builtin(__builtin_amdgcn_s_wait_tensorcnt)
#define HAVE_TDM 1
#endif

// ---------------------------------------------------------------------------
// WMMA fragment builders (CDNA5 v_wmma_f32_16x16x32_f16 layouts, wave32)
// A (16x32 f16): lane L -> row m = L&15, half = L>>4.
//   VGPR v holds K = (v>>2)*16 + half*8 + (v&3)*2 (+1 in high half)
// B (32x16 f16): lane L -> col n = L&15, half = L>>4.
//   VGPR v holds K = (v>>2)*16 + half*8 + (v&3)*2 (+1)
// LDS tiles stored K-contiguous so each VGPR is one aligned b32 load.
// ---------------------------------------------------------------------------
__device__ inline v16h frag_A_lds(const _Float16* base, int ld, int lane, int koff) {
  int m = lane & 15, kh = lane >> 4;
  v16h a;
#pragma unroll
  for (int v = 0; v < 8; ++v) {
    int k = koff + ((v >> 2) << 4) + (kh << 3) + ((v & 3) << 1);
    ((h2*)&a)[v] = *(const h2*)(base + m * ld + k);
  }
  return a;
}

__device__ inline v16h frag_B_lds(const _Float16* base, int ld, int lane, int koff) {
  int n = lane & 15, kh = lane >> 4;
  v16h b;
#pragma unroll
  for (int v = 0; v < 8; ++v) {
    int k = koff + ((v >> 2) << 4) + (kh << 3) + ((v & 3) << 1);
    ((h2*)&b)[v] = *(const h2*)(base + n * ld + k);
  }
  return b;
}

// ---------------------------------------------------------------------------
// Tensor Data Mover: DMA a [rows x cols] f16 tile (row-major, packed) from
// global memory into LDS. D# built per CDNA5 ISA 8.3/8.4.
// ---------------------------------------------------------------------------
__device__ inline void tdm_load_rows_f16(const void* gsrc, unsigned lds_byte_off,
                                         unsigned rows, unsigned cols) {
#ifdef HAVE_TDM
  unsigned long long ga = (unsigned long long)gsrc;
  v4u g0;
  g0[0] = 1u;                                   // count=1 (valid), user-mode, no gather
  g0[1] = lds_byte_off;                         // lds_addr (bytes)
  g0[2] = (unsigned)ga;                         // global_addr[31:0]
  g0[3] = (unsigned)((ga >> 32) & 0x01FFFFFFu) | (2u << 30); // addr[56:32] | type=2
  v8i g1;
  g1[0] = (int)(1u << 16);                      // data_size=1 -> 2 bytes/elem
  g1[1] = (int)((cols & 0xFFFFu) << 16);        // tensor_dim0[15:0]
  g1[2] = (int)(((cols >> 16) & 0xFFFFu) | ((rows & 0xFFFFu) << 16)); // dim0 hi | dim1 lo
  g1[3] = (int)(((rows >> 16) & 0xFFFFu) | ((cols & 0xFFFFu) << 16)); // dim1 hi | tile_dim0
  g1[4] = (int)(rows & 0xFFFFu);                // tile_dim1 | tile_dim2=0
  g1[5] = (int)cols;                            // tensor_dim0_stride[31:0]
  g1[6] = 0;                                    // stride0 hi | stride1 lo (2D: unused)
  g1[7] = 0;
  v4i z4 = {0, 0, 0, 0};
#if defined(__clang_major__) && (__clang_major__ >= 23)
  v8i z8 = {0, 0, 0, 0, 0, 0, 0, 0};
  __builtin_amdgcn_tensor_load_to_lds(g0, g1, z4, z4, z8, 0);
#else
  __builtin_amdgcn_tensor_load_to_lds(g0, g1, z4, z4, 0);
#endif
#else
  (void)gsrc; (void)lds_byte_off; (void)rows; (void)cols;
#endif
}

// ---------------------------------------------------------------------------
// Simple elementwise stages
// ---------------------------------------------------------------------------
__global__ void avgpool2_f32(const float* __restrict__ in, float* __restrict__ out,
                             int BC, int H, int W) {
  int Ho = H >> 1, Wo = W >> 1;
  int total = BC * Ho * Wo;
  int gid = blockIdx.x * blockDim.x + threadIdx.x;
  if (gid >= total) return;
  int x = gid % Wo; int t = gid / Wo; int y = t % Ho; int bc = t / Ho;
  const float* p = in + ((size_t)bc * H + 2 * y) * W + 2 * x;
  out[gid] = 0.25f * (p[0] + p[1] + p[W] + p[W + 1]);
}

__global__ void maxpool2_f16(const _Float16* __restrict__ in, _Float16* __restrict__ out,
                             int BC, int H, int W) {
  int Ho = H >> 1, Wo = W >> 1;
  int total = BC * Ho * Wo;
  int gid = blockIdx.x * blockDim.x + threadIdx.x;
  if (gid >= total) return;
  int x = gid % Wo; int t = gid / Wo; int y = t % Ho; int bc = t / Ho;
  const _Float16* p = in + ((size_t)bc * H + 2 * y) * W + 2 * x;
  float m = fmaxf(fmaxf((float)p[0], (float)p[1]), fmaxf((float)p[W], (float)p[W + 1]));
  out[gid] = (_Float16)m;
}

// conv1: 3->64 ch, 3x3 pad1, relu. K=27 too small for WMMA; direct VALU (<2% FLOPs).
__global__ void conv1_direct(const float* __restrict__ in, const float* __restrict__ w,
                             const float* __restrict__ bias, _Float16* __restrict__ out,
                             int B, int H, int W) {
  int total = B * 64 * H * W;
  int gid = blockIdx.x * blockDim.x + threadIdx.x;
  if (gid >= total) return;
  int x = gid % W; int t = gid / W; int y = t % H; t /= H; int oc = t % 64; int b = t / 64;
  float s = bias[oc];
#pragma unroll
  for (int ic = 0; ic < 3; ++ic)
#pragma unroll
    for (int ky = 0; ky < 3; ++ky) {
      int iy = y + ky - 1;
      if (iy < 0 || iy >= H) continue;
#pragma unroll
      for (int kx = 0; kx < 3; ++kx) {
        int ix = x + kx - 1;
        if (ix < 0 || ix >= W) continue;
        s += in[((size_t)(b * 3 + ic) * H + iy) * W + ix] * w[((oc * 3 + ic) * 3 + ky) * 3 + kx];
      }
    }
  out[gid] = (_Float16)fmaxf(s, 0.f);
}

// ---------------------------------------------------------------------------
// 3x3 conv as implicit GEMM via WMMA: tile = 16 out-channels x 64 pixels,
// K = Cin*9 in chunks of 32. 128 threads = 4 waves, one 16x16 subtile each.
// ---------------------------------------------------------------------------
__global__ __launch_bounds__(128) void conv3x3_wmma(
    const _Float16* __restrict__ in, const float* __restrict__ w,
    const float* __restrict__ bias, _Float16* __restrict__ out,
    int B, int Cin, int Cout, int H, int W) {
  const int tid = threadIdx.x, lane = tid & 31, wave = tid >> 5;
  const int x0 = blockIdx.x * 64;
  const int by = blockIdx.y;
  const int b = by / H, y = by - b * H;
  const int oc0 = blockIdx.z * 16;
  const int K = Cin * 9;

  __shared__ _Float16 sA[16 * 34];
  __shared__ _Float16 sB[64 * 34];
  __shared__ int kic[32], kky[32], kkx[32];

  v8f acc = {0.f, 0.f, 0.f, 0.f, 0.f, 0.f, 0.f, 0.f};

  for (int k0 = 0; k0 < K; k0 += 32) {
    if (tid < 32) {
      int k = k0 + tid;
      int ic = k / 9, r = k - ic * 9;
      int ky = r / 3;
      kic[tid] = ic; kky[tid] = ky; kkx[tid] = r - 3 * ky;
    }
    __syncthreads();
    // A tile: weights [16 oc][32 k], coalesced along k
#pragma unroll
    for (int j = 0; j < 4; ++j) {
      int e = j * 128 + tid;
      int kk = e & 31, m = e >> 5;
      int k = k0 + kk;
      sA[m * 34 + kk] = (k < K) ? (_Float16)w[(size_t)(oc0 + m) * K + k] : (_Float16)0.f;
    }
    // B tile: im2col patches [64 n][32 k]; 64 consecutive lanes share kk -> coalesced
#pragma unroll 4
    for (int j = 0; j < 16; ++j) {
      int kk = (j << 1) + (tid >> 6);
      int n = tid & 63;
      int k = k0 + kk;
      _Float16 v = (_Float16)0.f;
      if (k < K) {
        int iy = y + kky[kk] - 1;
        int ix = x0 + n + kkx[kk] - 1;
        if (iy >= 0 && iy < H && ix >= 0 && ix < W) {
          size_t base = ((size_t)(b * Cin + kic[kk]) * H + iy) * W + ix;
          v = in[base];
          // prefetch ~next K-chunk's channel plane (k advances 32 -> ~3.5 channels)
          if (k + 32 < K && kic[kk] + 3 < Cin)
            __builtin_prefetch(&in[base + (size_t)3 * H * W], 0, 1);
        }
      }
      sB[n * 34 + kk] = v;
    }
    __syncthreads();
    v16h af = frag_A_lds(sA, 34, lane, 0);
    v16h bf = frag_B_lds(sB + wave * 16 * 34, 34, lane, 0);
    acc = __builtin_amdgcn_wmma_f32_16x16x32_f16(false, af, false, bf, (short)0, acc, false, false);
    __syncthreads();
  }

  int x = x0 + wave * 16 + (lane & 15);
  if (x < W) {
    int half = lane >> 4;
#pragma unroll
    for (int r = 0; r < 8; ++r) {
      int oc = oc0 + r + (half << 3);
      float v = acc[r] + bias[oc];
      out[((size_t)(b * Cout + oc) * H + y) * W + x] = (_Float16)fmaxf(v, 0.f);
    }
  }
}

// ---------------------------------------------------------------------------
// 1x1 "match" conv (128->16) + LeakyReLU(0.2), WMMA, f32 output
// ---------------------------------------------------------------------------
__global__ __launch_bounds__(128) void conv1x1_wmma_leaky(
    const _Float16* __restrict__ in, const float* __restrict__ w,
    const float* __restrict__ bias, float* __restrict__ out,
    int B, int Cin, int Cout, int L) {
  const int tid = threadIdx.x, lane = tid & 31, wave = tid >> 5;
  const int l0 = blockIdx.x * 64;
  const int b = blockIdx.y;

  __shared__ _Float16 sA[16 * 34];
  __shared__ _Float16 sB[64 * 34];

  v8f acc = {0.f, 0.f, 0.f, 0.f, 0.f, 0.f, 0.f, 0.f};

  for (int k0 = 0; k0 < Cin; k0 += 32) {
#pragma unroll
    for (int j = 0; j < 4; ++j) {
      int e = j * 128 + tid;
      int kk = e & 31, m = e >> 5;
      sA[m * 34 + kk] = (_Float16)w[(size_t)m * Cin + k0 + kk];
    }
#pragma unroll 4
    for (int j = 0; j < 16; ++j) {
      int kk = (j << 1) + (tid >> 6);
      int n = tid & 63;
      int l = l0 + n;
      sB[n * 34 + kk] = (l < L) ? in[((size_t)b * Cin + k0 + kk) * L + l] : (_Float16)0.f;
    }
    __syncthreads();
    v16h af = frag_A_lds(sA, 34, lane, 0);
    v16h bf = frag_B_lds(sB + wave * 16 * 34, 34, lane, 0);
    acc = __builtin_amdgcn_wmma_f32_16x16x32_f16(false, af, false, bf, (short)0, acc, false, false);
    __syncthreads();
  }

  int l = l0 + wave * 16 + (lane & 15);
  if (l < L) {
    int half = lane >> 4;
#pragma unroll
    for (int r = 0; r < 8; ++r) {
      int oc = r + (half << 3);
      float v = acc[r] + bias[oc];
      v = v > 0.f ? v : 0.2f * v;
      out[((size_t)b * Cout + oc) * L + l] = v;
    }
  }
}

// ---------------------------------------------------------------------------
// Unfold 3x3 (reflect pad 1) + L2 normalize
// ---------------------------------------------------------------------------
__device__ inline int refl(int i, int n) { return i < 0 ? -i : (i >= n ? 2 * n - 2 - i : i); }

// query patches: qn[b][f(0..159)][l], K-major (B-matrix feed), pad rows 144..159 = 0
__global__ void unfold_norm_q(const float* __restrict__ f, _Float16* __restrict__ qn,
                              int H, int W) {
  int L = H * W;
  int l = blockIdx.x * blockDim.x + threadIdx.x;
  if (l >= L) return;
  int b = blockIdx.y;
  int y = l / W, x = l - y * W;
  float ss = 0.f;
  for (int c = 0; c < 16; ++c)
#pragma unroll
    for (int t = 0; t < 9; ++t) {
      int iy = refl(y + t / 3 - 1, H), ix = refl(x + t % 3 - 1, W);
      float v = f[((size_t)(b * 16 + c) * H + iy) * W + ix];
      ss += v * v;
    }
  float inv = 1.f / fmaxf(sqrtf(ss), 1e-12f);
  for (int c = 0; c < 16; ++c)
#pragma unroll
    for (int t = 0; t < 9; ++t) {
      int iy = refl(y + t / 3 - 1, H), ix = refl(x + t % 3 - 1, W);
      float v = f[((size_t)(b * 16 + c) * H + iy) * W + ix];
      qn[((size_t)b * 160 + c * 9 + t) * L + l] = (_Float16)(v * inv);
    }
  for (int ff = 144; ff < 160; ++ff)
    qn[((size_t)b * 160 + ff) * L + l] = (_Float16)0.f;
}

// key patches: wn[b][l][f(0..159)], row-major (A-matrix feed), pad 144..159 = 0
__global__ void unfold_norm_k(const float* __restrict__ f, _Float16* __restrict__ wn,
                              int H, int W) {
  int L = H * W;
  int l = blockIdx.x * blockDim.x + threadIdx.x;
  if (l >= L) return;
  int b = blockIdx.y;
  int y = l / W, x = l - y * W;
  float ss = 0.f;
  for (int c = 0; c < 16; ++c)
#pragma unroll
    for (int t = 0; t < 9; ++t) {
      int iy = refl(y + t / 3 - 1, H), ix = refl(x + t % 3 - 1, W);
      float v = f[((size_t)(b * 16 + c) * H + iy) * W + ix];
      ss += v * v;
    }
  float inv = 1.f / fmaxf(sqrtf(ss), 1e-12f);
  _Float16* row = wn + ((size_t)b * L + l) * 160;
  for (int c = 0; c < 16; ++c)
#pragma unroll
    for (int t = 0; t < 9; ++t) {
      int iy = refl(y + t / 3 - 1, H), ix = refl(x + t % 3 - 1, W);
      float v = f[((size_t)(b * 16 + c) * H + iy) * W + ix];
      row[c * 9 + t] = (_Float16)(v * inv);
    }
  for (int ff = 144; ff < 160; ++ff) row[ff] = (_Float16)0.f;
}

// ---------------------------------------------------------------------------
// Cosine-similarity GEMM [Lk,160]x[160,Lq] with fused max/argmax over Lk.
// Block: 4 waves, 64 query columns cached in LDS once. Key tiles (32x160 f16)
// are streamed by the Tensor Data Mover into a double buffer, pipelined one
// tile ahead of the 10-WMMA compute stage; sync via s_wait_tensorcnt.
// ---------------------------------------------------------------------------
__global__ __launch_bounds__(128) void cosine_sim_argmax_wmma(
    const _Float16* __restrict__ qn,  // [B][160][Lq]
    const _Float16* __restrict__ wn,  // [B][Lk][160]
    float* __restrict__ rel, int* __restrict__ idx, int Lq, int Lk) {
  const int tid = threadIdx.x, lane = tid & 31, wave = tid >> 5;
  const int b = blockIdx.y;
  const int l0 = blockIdx.x * 64;
  const int half = lane >> 4;

  __shared__ _Float16 sQ[64 * 168];      // query tile, n-major, k-contiguous (+pad)
  __shared__ _Float16 sW[2][32 * 160];   // double-buffered key tiles (TDM, packed)

  // stage 64x160 query tile (transposed); coalesced on l
  for (int e = tid; e < 64 * 160; e += 128) {
    int k = e >> 6, n = e & 63;
    sQ[n * 168 + k] = qn[((size_t)b * 160 + k) * Lq + l0 + n];
  }

  float best = -1e30f;
  int bidx = 0;
  const _Float16* sQw = sQ + wave * 16 * 168;
  const _Float16* wbase = wn + (size_t)b * Lk * 160;
  const int nTiles = Lk / 32;

#ifdef HAVE_TDM
  const unsigned ldsW0 = (unsigned)(unsigned long long)(const void*)&sW[0][0];
  const unsigned ldsW1 = (unsigned)(unsigned long long)(const void*)&sW[1][0];
  if (wave == 0)
    tdm_load_rows_f16(wbase, ldsW0, 32u, 160u);

  for (int i = 0; i < nTiles; ++i) {
    if (wave == 0) {
      if (i + 1 < nTiles) {
        tdm_load_rows_f16(wbase + (size_t)(i + 1) * 32 * 160,
                          ((i + 1) & 1) ? ldsW1 : ldsW0, 32u, 160u);
        __builtin_amdgcn_s_wait_tensorcnt(1);  // tile i landed
      } else {
        __builtin_amdgcn_s_wait_tensorcnt(0);
      }
    }
    __syncthreads();
    const _Float16* sWc = sW[i & 1];
    v8f acc0 = {0.f, 0.f, 0.f, 0.f, 0.f, 0.f, 0.f, 0.f};
    v8f acc1 = {0.f, 0.f, 0.f, 0.f, 0.f, 0.f, 0.f, 0.f};
#pragma unroll
    for (int c = 0; c < 5; ++c) {
      v16h bf = frag_B_lds(sQw, 168, lane, c * 32);
      v16h a0 = frag_A_lds(sWc, 160, lane, c * 32);
      v16h a1 = frag_A_lds(sWc + 16 * 160, 160, lane, c * 32);
      acc0 = __builtin_amdgcn_wmma_f32_16x16x32_f16(false, a0, false, bf, (short)0, acc0, false, false);
      acc1 = __builtin_amdgcn_wmma_f32_16x16x32_f16(false, a1, false, bf, (short)0, acc1, false, false);
    }
    int m0 = i * 32;
#pragma unroll
    for (int r = 0; r < 8; ++r) {
      float v = acc0[r];
      if (v > best) { best = v; bidx = m0 + r + (half << 3); }
    }
#pragma unroll
    for (int r = 0; r < 8; ++r) {
      float v = acc1[r];
      if (v > best) { best = v; bidx = m0 + 16 + r + (half << 3); }
    }
    __syncthreads();  // all reads of sW[i&1] done before wave0 refills it
  }
#else
  for (int i = 0; i < nTiles; ++i) {
    __syncthreads();
    for (int e = tid; e < 32 * 160; e += 128)
      sW[0][e] = wbase[(size_t)i * 32 * 160 + e];
    __syncthreads();
    v8f acc0 = {0.f, 0.f, 0.f, 0.f, 0.f, 0.f, 0.f, 0.f};
    v8f acc1 = {0.f, 0.f, 0.f, 0.f, 0.f, 0.f, 0.f, 0.f};
#pragma unroll
    for (int c = 0; c < 5; ++c) {
      v16h bf = frag_B_lds(sQw, 168, lane, c * 32);
      v16h a0 = frag_A_lds(sW[0], 160, lane, c * 32);
      v16h a1 = frag_A_lds(sW[0] + 16 * 160, 160, lane, c * 32);
      acc0 = __builtin_amdgcn_wmma_f32_16x16x32_f16(false, a0, false, bf, (short)0, acc0, false, false);
      acc1 = __builtin_amdgcn_wmma_f32_16x16x32_f16(false, a1, false, bf, (short)0, acc1, false, false);
    }
    int m0 = i * 32;
#pragma unroll
    for (int r = 0; r < 8; ++r) {
      float v = acc0[r];
      if (v > best) { best = v; bidx = m0 + r + (half << 3); }
    }
#pragma unroll
    for (int r = 0; r < 8; ++r) {
      float v = acc1[r];
      if (v > best) { best = v; bidx = m0 + 16 + r + (half << 3); }
    }
  }
#endif

  // combine the two lane-halves holding the same query column
  float ob = __shfl_xor(best, 16, 32);
  int oi = __shfl_xor(bidx, 16, 32);
  if (ob > best || (ob == best && oi < bidx)) { best = ob; bidx = oi; }
  if (half == 0) {
    int l = l0 + wave * 16 + lane;
    rel[(size_t)b * Lq + l] = best;
    idx[(size_t)b * Lq + l] = bidx;
  }
}

// ---------------------------------------------------------------------------
// Host-side orchestration
// ---------------------------------------------------------------------------
extern "C" void kernel_launch(void* const* d_in, const int* in_sizes, int n_in,
                              void* d_out, int out_size, void* d_ws, size_t ws_size,
                              hipStream_t stream) {
  const float* query = (const float*)d_in[0];
  const float* key   = (const float*)d_in[1];
  const float* w1 = (const float*)d_in[2];
  const float* b1 = (const float*)d_in[3];
  const float* w2 = (const float*)d_in[4];
  const float* b2 = (const float*)d_in[5];
  const float* w3 = (const float*)d_in[6];
  const float* b3 = (const float*)d_in[7];
  const float* wm = (const float*)d_in[8];
  const float* bm = (const float*)d_in[9];
  (void)in_sizes; (void)n_in; (void)out_size; (void)ws_size;

  const int B = 2;
  const int Lq = 96 * 96;   // 9216
  const int Lk = 48 * 48;   // 2304

  char* ws = (char*)d_ws;
  size_t off = 0;
  auto alloc = [&](size_t bytes) -> void* {
    void* p = ws + off;
    off = (off + bytes + 255) & ~(size_t)255;
    return p;
  };

  float*    kpool = (float*)alloc((size_t)B * 3 * 96 * 96 * 4);
  _Float16* q1    = (_Float16*)alloc((size_t)B * 64 * 192 * 192 * 2);
  _Float16* k1    = (_Float16*)alloc((size_t)B * 64 * 96 * 96 * 2);
  _Float16* q2    = (_Float16*)alloc((size_t)B * 64 * 192 * 192 * 2);
  _Float16* k2    = (_Float16*)alloc((size_t)B * 64 * 96 * 96 * 2);
  _Float16* q2p   = (_Float16*)alloc((size_t)B * 64 * 96 * 96 * 2);
  _Float16* k2p   = (_Float16*)alloc((size_t)B * 64 * 48 * 48 * 2);
  _Float16* q3    = (_Float16*)alloc((size_t)B * 128 * 96 * 96 * 2);
  _Float16* k3    = (_Float16*)alloc((size_t)B * 128 * 48 * 48 * 2);
  float*    qf    = (float*)alloc((size_t)B * 16 * 96 * 96 * 4);
  float*    kf    = (float*)alloc((size_t)B * 16 * 48 * 48 * 4);
  _Float16* qn    = (_Float16*)alloc((size_t)B * 160 * Lq * 2);
  _Float16* wnb   = (_Float16*)alloc((size_t)B * Lk * 160 * 2);

  // 1. key avgpool 192 -> 96
  { int total = B * 3 * 96 * 96;
    avgpool2_f32<<<(total + 255) / 256, 256, 0, stream>>>(key, kpool, B * 3, 192, 192); }

  // 2. conv1 (3->64, relu) query @192, key @96
  { int total = B * 64 * 192 * 192;
    conv1_direct<<<(total + 255) / 256, 256, 0, stream>>>(query, w1, b1, q1, B, 192, 192); }
  { int total = B * 64 * 96 * 96;
    conv1_direct<<<(total + 255) / 256, 256, 0, stream>>>(kpool, w1, b1, k1, B, 96, 96); }

  // 3. conv2 (64->64, relu) — WMMA implicit GEMM
  conv3x3_wmma<<<dim3(3, B * 192, 4), 128, 0, stream>>>(q1, w2, b2, q2, B, 64, 64, 192, 192);
  conv3x3_wmma<<<dim3(2, B * 96, 4), 128, 0, stream>>>(k1, w2, b2, k2, B, 64, 64, 96, 96);

  // 4. maxpool 2x2
  { int total = B * 64 * 96 * 96;
    maxpool2_f16<<<(total + 255) / 256, 256, 0, stream>>>(q2, q2p, B * 64, 192, 192); }
  { int total = B * 64 * 48 * 48;
    maxpool2_f16<<<(total + 255) / 256, 256, 0, stream>>>(k2, k2p, B * 64, 96, 96); }

  // 5. conv3 (64->128, relu) — WMMA implicit GEMM
  conv3x3_wmma<<<dim3(2, B * 96, 8), 128, 0, stream>>>(q2p, w3, b3, q3, B, 64, 128, 96, 96);
  conv3x3_wmma<<<dim3(1, B * 48, 8), 128, 0, stream>>>(k2p, w3, b3, k3, B, 64, 128, 48, 48);

  // 6. match 1x1 (128->16) + LeakyReLU(0.2) — WMMA
  conv1x1_wmma_leaky<<<dim3((Lq + 63) / 64, B), 128, 0, stream>>>(q3, wm, bm, qf, B, 128, 16, Lq);
  conv1x1_wmma_leaky<<<dim3((Lk + 63) / 64, B), 128, 0, stream>>>(k3, wm, bm, kf, B, 128, 16, Lk);

  // 7. unfold(3x3, reflect) + L2-normalize, pad K 144->160
  unfold_norm_q<<<dim3((Lq + 255) / 256, B), 256, 0, stream>>>(qf, qn, 96, 96);
  unfold_norm_k<<<dim3((Lk + 255) / 256, B), 256, 0, stream>>>(kf, wnb, 48, 48);

  // 8. cosine-sim GEMM + fused max/argmax over Lk (TDM-streamed key tiles)
  float* rel = (float*)d_out;                    // [B,1,96,96]
  int*   hid = (int*)(rel + (size_t)B * Lq);     // [B,Lq]
  cosine_sim_argmax_wmma<<<dim3(Lq / 64, B), 128, 0, stream>>>(qn, wnb, rel, hid, Lq, Lk);
}
